// RelMultiHeadAttention_90142773608751
// MI455X (gfx1250) — matrix operand order repaired and checked by reading, not compiled, not run on hardware
//
#include <hip/hip_runtime.h>
#include <hip/hip_bf16.h>
#include <stdint.h>

// ---------------- problem constants ----------------
#define DM     1024
#define NHEAD  16
#define DEPTH  64
#define BATCH  2
#define QLEN   1024
#define MLEN   1024
#define KLEN   2048      // MLEN + QLEN
#define RPAD   2112      // Rp rows padded (max diagonal read = 2079)

typedef __attribute__((ext_vector_type(16))) _Float16 v16h;
typedef __attribute__((ext_vector_type(8)))  float    v8f;

union FragU { v16h v; uint32_t u[8]; };
union HBits { _Float16 h; uint16_t u; };

__device__ __forceinline__ uint16_t f2h_bits(float f) { HBits c; c.h = (_Float16)f; return c.u; }
__device__ __forceinline__ float    h2f_bits(uint16_t u) { HBits c; c.u = u; return (float)c.h; }

__device__ __forceinline__ v8f vzero8() { v8f z = {0.f,0.f,0.f,0.f,0.f,0.f,0.f,0.f}; return z; }

__device__ __forceinline__ v8f wmma16(v16h a, v16h b, v8f c) {
  // D = A(16x32 f16) * B(32x16 f16) + C(16x16 f32)
  return __builtin_amdgcn_wmma_f32_16x16x32_f16(false, a, false, b, (short)0, c, false, false);
}

// ---- A fragment: 16x32 f16, source row-major (K contiguous), contiguous k order (not the ISA table order):
// lane(l): m=l&15, hi=l>>4; vgpr i -> K=2i+16hi (pairs contiguous)
__device__ __forceinline__ v16h load_afrag(const uint16_t* base, int ld, int lane) {
  int m = lane & 15, hi = lane >> 4;
  const uint16_t* rowp = base + (size_t)m * ld;
  FragU r;
#pragma unroll
  for (int i = 0; i < 8; ++i) {
    int kk = 2 * i + 16 * hi;   // even -> 4B aligned pair
    r.u[i] = *(const uint32_t*)(rowp + kk);
  }
  return r.v;
}

// A fragment built from f16 row + f32 additive vector (Q+u / Q+v), re-rounded to f16
__device__ __forceinline__ v16h load_afrag_add(const uint16_t* base, int ld,
                                               const float* addv, int lane) {
  int m = lane & 15, hi = lane >> 4;
  const uint16_t* rowp = base + (size_t)m * ld;
  FragU r;
#pragma unroll
  for (int i = 0; i < 8; ++i) {
    int kk = 2 * i + 16 * hi;
    uint32_t lo = f2h_bits(h2f_bits(rowp[kk])     + addv[kk]);
    uint32_t hh = f2h_bits(h2f_bits(rowp[kk + 1]) + addv[kk + 1]);
    r.u[i] = lo | (hh << 16);
  }
  return r.v;
}

// ---- B fragment: 32x16 f16 where source is laid out (N rows x K cols, K contiguous)
// lane(l): n=l&15, hi=l>>4; vgpr i -> K=2i+16hi (pairs contiguous)
__device__ __forceinline__ v16h load_bfrag(const uint16_t* base, int ld, int lane) {
  int n = lane & 15, hi = lane >> 4;
  const uint16_t* rowp = base + (size_t)n * ld + 16 * hi;
  FragU r;
#pragma unroll
  for (int i = 0; i < 8; ++i) r.u[i] = *(const uint32_t*)(rowp + 2 * i);
  return r.v;
}

// B fragment where source is (K rows x N cols) row-major (used for V: keys x depth)
__device__ __forceinline__ v16h load_bfrag_strided(const uint16_t* base, int ld, int lane) {
  int n = lane & 15, hi = lane >> 4;
  const uint16_t* p = base + n + (size_t)(16 * hi) * ld;
  FragU r;
#pragma unroll
  for (int i = 0; i < 8; ++i) {
    uint32_t lo = p[(size_t)(2 * i) * ld];
    uint32_t hh = p[(size_t)(2 * i + 1) * ld];
    r.u[i] = lo | (hh << 16);
  }
  return r.v;
}

// A fragment from a 16x32 f16 tile in LDS (row stride 32)
__device__ __forceinline__ v16h load_afrag_lds(const uint16_t* base, int lane) {
  int m = lane & 15, hi = lane >> 4;
  const uint16_t* rowp = base + m * 32;
  FragU r;
#pragma unroll
  for (int i = 0; i < 8; ++i) {
    int kk = 2 * i + 16 * hi;
    r.u[i] = *(const uint32_t*)(rowp + kk);
  }
  return r.v;
}

// ---------------- prep kernels ----------------
__global__ void k_build_xa(const float* __restrict__ x, const float* __restrict__ xm,
                           uint16_t* __restrict__ xa) {
  int idx = blockIdx.x * blockDim.x + threadIdx.x;       // BATCH*KLEN*DM
  int d = idx & (DM - 1);
  int t = (idx >> 10) & (KLEN - 1);
  int b = idx >> 21;
  float val = (t < MLEN) ? xm[((size_t)(b * MLEN + t)) * DM + d]
                         : x[((size_t)(b * QLEN + (t - MLEN))) * DM + d];
  xa[idx] = f2h_bits(val);
}

__global__ void k_transpose(const float* __restrict__ W, uint16_t* __restrict__ Wt,
                            int K, int N) {  // W: KxN -> Wt: NxK (f16)
  int idx = blockIdx.x * blockDim.x + threadIdx.x;
  if (idx >= K * N) return;
  int k = idx % K;
  int n = idx / K;
  Wt[(size_t)n * K + k] = f2h_bits(W[(size_t)k * N + n]);
}

__global__ void k_gen_R(uint16_t* __restrict__ R) {  // RPAD x DM, rows>=KLEN zero
  int idx = blockIdx.x * blockDim.x + threadIdx.x;     // RPAD*DM
  int d = idx & (DM - 1);
  int p = idx >> 10;
  float val = 0.f;
  if (p < KLEN) {
    int j = (DM - 1) - d;                 // reversed last axis
    double expo = (double)(j & ~1) / (double)DM;
    double freq = exp(expo * log(1.0e-4));
    double ang  = (double)p * freq;
    val = (float)((j & 1) ? sin(ang) : cos(ang));
  }
  R[idx] = f2h_bits(val);
}

// ---------------- generic WMMA GEMM: C = A(MxK f16) * Bt(NxK f16)^T + bias ----------------
// MODE 0: f16 output, MODE 1: f32 output with ReLU
template <int MODE>
__global__ __launch_bounds__(256) void gemm_f16_kernel(
    const uint16_t* __restrict__ A, int lda, const uint16_t* __restrict__ Bt, int ldb,
    const float* __restrict__ bias, void* __restrict__ outp, int ldo, int M, int N, int K) {
  int lane = threadIdx.x & 31;
  int wv   = threadIdx.x >> 5;
  int nt   = N >> 6;                       // 64-wide N tiles per wave
  int wid  = blockIdx.x * 8 + wv;
  if (wid >= (M >> 4) * nt) return;
  int m0 = (wid / nt) * 16;
  int n0 = (wid % nt) * 64;

  v8f acc0 = vzero8(), acc1 = vzero8(), acc2 = vzero8(), acc3 = vzero8();
  for (int k0 = 0; k0 < K; k0 += 32) {
    v16h a = load_afrag(A + (size_t)m0 * lda + k0, lda, lane);
    acc0 = wmma16(a, load_bfrag(Bt + (size_t)(n0 +  0) * ldb + k0, ldb, lane), acc0);
    acc1 = wmma16(a, load_bfrag(Bt + (size_t)(n0 + 16) * ldb + k0, ldb, lane), acc1);
    acc2 = wmma16(a, load_bfrag(Bt + (size_t)(n0 + 32) * ldb + k0, ldb, lane), acc2);
    acc3 = wmma16(a, load_bfrag(Bt + (size_t)(n0 + 48) * ldb + k0, ldb, lane), acc3);
  }
  int jj = lane & 15, hi = lane >> 4;
#pragma unroll
  for (int r = 0; r < 8; ++r) {
    int m = m0 + r + 8 * hi;
    float v0 = acc0[r] + bias[n0 +  0 + jj];
    float v1 = acc1[r] + bias[n0 + 16 + jj];
    float v2 = acc2[r] + bias[n0 + 32 + jj];
    float v3 = acc3[r] + bias[n0 + 48 + jj];
    if (MODE == 1) {
      float* o = (float*)outp;
      o[(size_t)m * ldo + n0 +  0 + jj] = fmaxf(v0, 0.f);
      o[(size_t)m * ldo + n0 + 16 + jj] = fmaxf(v1, 0.f);
      o[(size_t)m * ldo + n0 + 32 + jj] = fmaxf(v2, 0.f);
      o[(size_t)m * ldo + n0 + 48 + jj] = fmaxf(v3, 0.f);
    } else {
      uint16_t* o = (uint16_t*)outp;
      o[(size_t)m * ldo + n0 +  0 + jj] = f2h_bits(v0);
      o[(size_t)m * ldo + n0 + 16 + jj] = f2h_bits(v1);
      o[(size_t)m * ldo + n0 + 32 + jj] = f2h_bits(v2);
      o[(size_t)m * ldo + n0 + 48 + jj] = f2h_bits(v3);
    }
  }
}

// ---------------- fused rel-pos flash attention ----------------
// One wave owns a (b, h, 16-query) tile; streams keys 32 at a time.
// BD left-shift realized as per-row lane rotation over 3 diagonal Rp tiles.
__global__ __launch_bounds__(128) void attn_kernel(
    const uint16_t* __restrict__ qvk,   // (B, KLEN, 3*DM) f16; cols: [Q | V | K]
    const uint16_t* __restrict__ rp,    // (RPAD, DM) f16
    const float* __restrict__ uvec, const float* __restrict__ vvec,  // (NHEAD*DEPTH)
    uint16_t* __restrict__ attn) {      // (B, QLEN, DM) f16
  __shared__ uint32_t pbuf32[4][256];   // 4 waves x (16x32 f16) P tiles, 4B aligned
  const int lane = threadIdx.x & 31;
  const int wv   = threadIdx.x >> 5;
  const int widx = blockIdx.x * 4 + wv;
  const int q0 = (widx & 63) << 4;
  const int h  = (widx >> 6) & (NHEAD - 1);
  const int b  = widx >> 10;
  const int jj = lane & 15;
  const int hi = lane >> 4;
  const int hibase = hi << 4;
  const int ldq = 3 * DM;

  const uint16_t* qbase = qvk + ((size_t)(b * KLEN + MLEN + q0)) * ldq + 0 * DM + h * DEPTH;
  const uint16_t* kbase = qvk + ((size_t)(b * KLEN)) * ldq + 2 * DM + h * DEPTH;
  const uint16_t* vbase = qvk + ((size_t)(b * KLEN)) * ldq + 1 * DM + h * DEPTH;
  const uint16_t* rbase = rp + h * DEPTH;

  // loop-invariant A fragments: (Q+u) and (Q+v), depth split 0..31 / 32..63
  v16h qu0 = load_afrag_add(qbase + 0,  ldq, uvec + h * DEPTH + 0,  lane);
  v16h qu1 = load_afrag_add(qbase + 32, ldq, uvec + h * DEPTH + 32, lane);
  v16h qv0 = load_afrag_add(qbase + 0,  ldq, vvec + h * DEPTH + 0,  lane);
  v16h qv1 = load_afrag_add(qbase + 32, ldq, vvec + h * DEPTH + 32, lane);

  v8f o0 = vzero8(), o1 = vzero8(), o2 = vzero8(), o3 = vzero8();
  float mrun[8], lrun[8];
#pragma unroll
  for (int r = 0; r < 8; ++r) { mrun[r] = -1e30f; lrun[r] = 0.f; }

  uint16_t* myP = (uint16_t*)&pbuf32[wv][0];
  const float inv_scale = 1.0f / 64.0f;     // logits = (AC+BD)/DEPTH
  const int jmax = MLEN + q0 + 15;          // last key any row of this tile can see

  for (int kb0 = 0; kb0 <= jmax; kb0 += 32) {
    // AC score tiles: keys [kb0,kb0+16) and [kb0+16,kb0+32)
    v8f sa = vzero8(), sb = vzero8();
    sa = wmma16(qu0, load_bfrag(kbase + (size_t)kb0 * ldq + 0,  ldq, lane), sa);
    sa = wmma16(qu1, load_bfrag(kbase + (size_t)kb0 * ldq + 32, ldq, lane), sa);
    sb = wmma16(qu0, load_bfrag(kbase + (size_t)(kb0 + 16) * ldq + 0,  ldq, lane), sb);
    sb = wmma16(qu1, load_bfrag(kbase + (size_t)(kb0 + 16) * ldq + 32, ldq, lane), sb);

    // BD raw tiles over the diagonal band of Rp rows [r0, r0+48)
    int r0 = kb0 + (QLEN - 1) - q0 - 15;
    v8f t0 = vzero8(), t1 = vzero8(), t2 = vzero8();
    t0 = wmma16(qv0, load_bfrag(rbase + (size_t)(r0 +  0) * DM + 0,  DM, lane), t0);
    t0 = wmma16(qv1, load_bfrag(rbase + (size_t)(r0 +  0) * DM + 32, DM, lane), t0);
    t1 = wmma16(qv0, load_bfrag(rbase + (size_t)(r0 + 16) * DM + 0,  DM, lane), t1);
    t1 = wmma16(qv1, load_bfrag(rbase + (size_t)(r0 + 16) * DM + 32, DM, lane), t1);
    t2 = wmma16(qv0, load_bfrag(rbase + (size_t)(r0 + 32) * DM + 0,  DM, lane), t2);
    t2 = wmma16(qv1, load_bfrag(rbase + (size_t)(r0 + 32) * DM + 32, DM, lane), t2);

#pragma unroll
    for (int r = 0; r < 8; ++r) {
      int qq = r + (hi << 3);               // row index in tile (C layout)
      int rr = 15 - qq + jj;                // shifted column within [0,31)
      int src = hibase + (rr & 15);
      float b0 = __shfl(t0[r], src, 32);
      float b1 = __shfl(t1[r], src, 32);
      float b2 = __shfl(t2[r], src, 32);
      float bda = (rr < 16) ? b0 : b1;      // BD for key tile a
      float bdb = (rr < 16) ? b1 : b2;      // BD for key tile b (+16 rows)
      float s_a = (sa[r] + bda) * inv_scale;
      float s_b = (sb[r] + bdb) * inv_scale;
      int qg = q0 + qq;
      if (kb0 + jj      > MLEN + qg) s_a = -1e30f;   // causal mask
      if (kb0 + 16 + jj > MLEN + qg) s_b = -1e30f;
      // online softmax (row spread across 16 lanes of this half-wave)
      float mx = fmaxf(s_a, s_b);
#pragma unroll
      for (int d2 = 1; d2 < 16; d2 <<= 1) mx = fmaxf(mx, __shfl_xor(mx, d2, 32));
      float mnew  = fmaxf(mrun[r], mx);
      float alpha = expf(mrun[r] - mnew);
      float ea = expf(s_a - mnew);
      float eb = expf(s_b - mnew);
      float rs = ea + eb;
#pragma unroll
      for (int d2 = 1; d2 < 16; d2 <<= 1) rs += __shfl_xor(rs, d2, 32);
      lrun[r] = lrun[r] * alpha + rs;
      mrun[r] = mnew;
      o0[r] *= alpha; o1[r] *= alpha; o2[r] *= alpha; o3[r] *= alpha;
      myP[qq * 32 + jj]      = f2h_bits(ea);   // stage P (C layout -> LDS)
      myP[qq * 32 + 16 + jj] = f2h_bits(eb);
    }
    __builtin_amdgcn_wave_barrier();           // keep LDS store -> load order
    v16h pf = load_afrag_lds(myP, lane);       // P as A fragment (16q x 32k)
    const uint16_t* vb = vbase + (size_t)kb0 * ldq;
    o0 = wmma16(pf, load_bfrag_strided(vb +  0, ldq, lane), o0);
    o1 = wmma16(pf, load_bfrag_strided(vb + 16, ldq, lane), o1);
    o2 = wmma16(pf, load_bfrag_strided(vb + 32, ldq, lane), o2);
    o3 = wmma16(pf, load_bfrag_strided(vb + 48, ldq, lane), o3);
    __builtin_amdgcn_wave_barrier();
  }

#pragma unroll
  for (int r = 0; r < 8; ++r) {
    float linv = 1.0f / lrun[r];
    int qg = q0 + r + (hi << 3);
    size_t rowoff = ((size_t)(b * QLEN + qg)) * DM + h * DEPTH + jj;
    attn[rowoff +  0] = f2h_bits(o0[r] * linv);
    attn[rowoff + 16] = f2h_bits(o1[r] * linv);
    attn[rowoff + 32] = f2h_bits(o2[r] * linv);
    attn[rowoff + 48] = f2h_bits(o3[r] * linv);
  }
}

// ---------------- launcher ----------------
extern "C" void kernel_launch(void* const* d_in, const int* in_sizes, int n_in,
                              void* d_out, int out_size, void* d_ws, size_t ws_size,
                              hipStream_t stream) {
  const float* x       = (const float*)d_in[0];
  const float* x_mem   = (const float*)d_in[1];
  const float* W_emb_k = (const float*)d_in[2];
  const float* W_emb_b = (const float*)d_in[3];
  const float* W_pos_k = (const float*)d_in[4];
  const float* W_pos_b = (const float*)d_in[5];
  const float* u       = (const float*)d_in[6];
  const float* v       = (const float*)d_in[7];
  const float* W_out_k = (const float*)d_in[8];
  const float* W_out_b = (const float*)d_in[9];
  float* out = (float*)d_out;

  uint8_t* ws = (uint8_t*)d_ws;
  size_t off = 0;
  auto alloc = [&](size_t bytes) -> void* {
    void* p = ws + off;
    off += (bytes + 255) & ~(size_t)255;
    return p;
  };
  uint16_t* xa    = (uint16_t*)alloc((size_t)BATCH * KLEN * DM * 2);     // concat(x_mem,x) f16
  uint16_t* WembT = (uint16_t*)alloc((size_t)3 * DM * DM * 2);           // (3DM x DM)
  uint16_t* WposT = (uint16_t*)alloc((size_t)DM * DM * 2);
  uint16_t* WoutT = (uint16_t*)alloc((size_t)DM * DM * 2);
  uint16_t* Rm    = (uint16_t*)alloc((size_t)RPAD * DM * 2);             // sinusoid table
  uint16_t* QVK   = (uint16_t*)alloc((size_t)BATCH * KLEN * 3 * DM * 2);
  uint16_t* Rp    = (uint16_t*)alloc((size_t)RPAD * DM * 2);
  uint16_t* attn  = (uint16_t*)alloc((size_t)BATCH * QLEN * DM * 2);
  (void)ws_size; (void)in_sizes; (void)n_in; (void)out_size;

  // prep
  k_build_xa<<<(BATCH * KLEN * DM) / 256, 256, 0, stream>>>(x, x_mem, xa);
  k_transpose<<<(DM * 3 * DM) / 256, 256, 0, stream>>>(W_emb_k, WembT, DM, 3 * DM);
  k_transpose<<<(DM * DM) / 256, 256, 0, stream>>>(W_pos_k, WposT, DM, DM);
  k_transpose<<<(DM * DM) / 256, 256, 0, stream>>>(W_out_k, WoutT, DM, DM);
  k_gen_R<<<(RPAD * DM) / 256, 256, 0, stream>>>(Rm);

  // QVK = xa @ W_emb + b   (M=4096, N=3072, K=1024)
  gemm_f16_kernel<0><<<((BATCH * KLEN / 16) * (3 * DM / 64)) / 8, 256, 0, stream>>>(
      xa, DM, WembT, DM, W_emb_b, QVK, 3 * DM, BATCH * KLEN, 3 * DM, DM);
  // Rp = R @ W_pos + b     (M=2112 incl. zero pad rows, N=1024, K=1024)
  gemm_f16_kernel<0><<<((RPAD / 16) * (DM / 64)) / 8, 256, 0, stream>>>(
      Rm, DM, WposT, DM, W_pos_b, Rp, DM, RPAD, DM, DM);

  // fused rel-pos attention: 2048 (b,h,q-tile) waves, 4 waves/block
  attn_kernel<<<(BATCH * NHEAD * (QLEN / 16)) / 4, 128, 0, stream>>>(QVK, Rp, u, v, attn);

  // out = relu(attn @ W_out + b) -> f32
  gemm_f16_kernel<1><<<((BATCH * QLEN / 16) * (DM / 64)) / 8, 256, 0, stream>>>(
      attn, DM, WoutT, DM, W_out_b, out, DM, BATCH * QLEN, DM, DM);
}
